// MainModule_10617159156025
// MI455X (gfx1250) — compile-verified, matrix operand order checked by reading, not run on hardware
//
#include <hip/hip_runtime.h>

#define B_     32
#define S_     512
#define L_     300
#define D_     768
#define NDIM_  5
#define H_     384
#define G4_    1536   // 4*H
#define T_     305    // L + NDIM
#define NHEAD_ 18
#define NCLS_  4
#define KTOP_  384
#define MAXSPAN_ 8
#define NEG_   (-1e9f)

typedef _Float16 h16 __attribute__((ext_vector_type(16)));
typedef _Float16 h8  __attribute__((ext_vector_type(8)));
typedef float    f8  __attribute__((ext_vector_type(8)));
typedef unsigned int u32x4 __attribute__((ext_vector_type(4)));
typedef int          i32x4 __attribute__((ext_vector_type(4)));
typedef int          i32x8 __attribute__((ext_vector_type(8)));

// ---------------------------------------------------------------------------
// WMMA helpers: 16x16x32 f16 -> f32, "NT" fragment layout (ISA 7.12.2):
// lanes 0-15 -> row=lane,   K = {k0..k0+7, k0+16..k0+23}
// lanes16-31 -> row=lane-16,K = {k0+8..k0+15, k0+24..k0+31}
// ---------------------------------------------------------------------------
__device__ __forceinline__ h16 load_frag_nt(const _Float16* base, int ld, int k0) {
  int lane = threadIdx.x & 31;
  int row  = lane & 15;
  int ksel = (lane >> 4) & 1;
  const _Float16* p = base + (size_t)row * ld + k0 + ksel * 8;
  h8 a = *(const h8*)(p);
  h8 b = *(const h8*)(p + 16);
  h16 r;
#pragma unroll
  for (int i = 0; i < 8; i++) { r[i] = a[i]; r[i + 8] = b[i]; }
  return r;
}

__device__ __forceinline__ f8 wmma_f16(h16 a, h16 b, f8 c) {
  return __builtin_amdgcn_wmma_f32_16x16x32_f16(false, a, false, b, (short)0, c,
                                                false, false);
}

__device__ __forceinline__ float sigmoidf_(float x) { return 1.f / (1.f + __expf(-x)); }

__device__ __forceinline__ void wait_tensor0() {
#if defined(__has_builtin) && __has_builtin(__builtin_amdgcn_s_wait_tensorcnt)
  __builtin_amdgcn_s_wait_tensorcnt(0);
#else
  asm volatile("s_wait_tensorcnt 0x0" ::: "memory");
#endif
}

// TDM: async copy of a 2-D f16 tile (tile_d0 x tile_d1, row stride in elements)
// from global memory into LDS at byte offset lds_off. Descriptor per ISA ch.8.
__device__ __forceinline__ void tdm_load_2d_f16(unsigned lds_off, const void* gaddr,
                                                unsigned tens_d0, unsigned tens_d1,
                                                unsigned tile_d0, unsigned tile_d1,
                                                unsigned long long row_stride_el) {
  unsigned long long ga = (unsigned long long)gaddr;
  u32x4 g0;
  g0[0] = 1u;                                   // count=1, user descriptor
  g0[1] = lds_off;                              // lds_addr (bytes)
  g0[2] = (unsigned)ga;                         // global_addr[31:0]
  g0[3] = (unsigned)(ga >> 32) | (2u << 30);    // global_addr[56:32] | type=2
  i32x8 g1;
  g1[0] = (int)(1u << 16);                      // data_size=1 (2 bytes)
  g1[1] = (int)((tens_d0 & 0xFFFFu) << 16);     // tensor_dim0[15:0]
  g1[2] = (int)((tens_d0 >> 16) | ((tens_d1 & 0xFFFFu) << 16));
  g1[3] = (int)((tens_d1 >> 16) | (tile_d0 << 16));
  g1[4] = (int)(tile_d1 & 0xFFFFu);             // tile_dim1 (tile_dim2=0)
  g1[5] = (int)(unsigned)(row_stride_el & 0xFFFFFFFFull);
  g1[6] = (int)(unsigned)((row_stride_el >> 32) & 0xFFFFull);
  g1[7] = 0;
  i32x4 z4 = {0, 0, 0, 0};
#if __clang_major__ >= 23
  i32x8 z8 = {0, 0, 0, 0, 0, 0, 0, 0};
  __builtin_amdgcn_tensor_load_to_lds(g0, g1, z4, z4, z8, 0);
#else
  __builtin_amdgcn_tensor_load_to_lds(g0, g1, z4, z4, 0);
#endif
}

// ---------------------------------------------------------------------------
// small prep kernels
// ---------------------------------------------------------------------------
__global__ void k_cvt_f16(const float* __restrict__ s, _Float16* __restrict__ d, int n) {
  int i = blockIdx.x * blockDim.x + threadIdx.x;
  if (i < n) d[i] = (_Float16)s[i];
}

__global__ void k_add2(const float* __restrict__ a, const float* __restrict__ b,
                       float* __restrict__ o, int n) {
  int i = blockIdx.x * blockDim.x + threadIdx.x;
  if (i < n) o[i] = a[i] + b[i];
}

__global__ void k_first(const int* __restrict__ wi, int* __restrict__ firstb,
                        int* __restrict__ brkw) {
  int b = threadIdx.x;
  if (b >= B_) return;
  int f = L_, w = 0;
  for (int j = 0; j < L_; j++) {
    int wid = wi[(b * L_ + j) * 3 + 0];
    int st  = wi[(b * L_ + j) * 3 + 1];
    int en  = wi[(b * L_ + j) * 3 + 2];
    if ((en < S_) && (en - st) <= 0) { f = j; w = wid; break; }
  }
  firstb[b] = f;
  brkw[b]   = w;
}

// ---------------------------------------------------------------------------
// weighted embedding: one block per (b,l), 256 threads, 3 D-elems/thread
// ---------------------------------------------------------------------------
__global__ void k_wemb(const float* __restrict__ ernie, const int* __restrict__ wi,
                       const float* __restrict__ emb, const int* __restrict__ firstb,
                       const int* __restrict__ brkw, float* __restrict__ fuse,
                       _Float16* __restrict__ fuse_h) {
  __shared__ float red[256][MAXSPAN_];
  __shared__ float sw[MAXSPAN_];
  int bl = blockIdx.x;
  int b = bl / L_, l = bl % L_;
  int tid = threadIdx.x;
  int wid = wi[bl * 3 + 0], st = wi[bl * 3 + 1], en = wi[bl * 3 + 2];
  int len = en - st;
  bool valid = en < S_;
  int frst = firstb[b];
  const float* eb = ernie + (size_t)b * S_ * D_;

  float ch[MAXSPAN_][3], wv[3];
#pragma unroll
  for (int k2 = 0; k2 < 3; k2++) wv[k2] = emb[(size_t)wid * D_ + tid + k2 * 256];

#pragma unroll
  for (int s = 0; s < MAXSPAN_; s++) {
    int pos = st + s;
    pos = pos < 0 ? 0 : (pos > S_ - 1 ? S_ - 1 : pos);
    float p = 0.f;
#pragma unroll
    for (int k2 = 0; k2 < 3; k2++) {
      float cv = eb[(size_t)pos * D_ + tid + k2 * 256];
      ch[s][k2] = cv;
      p += wv[k2] * cv;
    }
    red[tid][s] = p;
  }
  __syncthreads();
  for (int str = 128; str; str >>= 1) {
    if (tid < str)
#pragma unroll
      for (int s = 0; s < MAXSPAN_; s++) red[tid][s] += red[tid + str][s];
    __syncthreads();
  }
  if (tid == 0) {
    int mlen = len < 0 ? 0 : (len > MAXSPAN_ ? MAXSPAN_ : len);
    float sc[MAXSPAN_], mx = NEG_;
    for (int s = 0; s < MAXSPAN_; s++) {
      sc[s] = (s < mlen) ? red[0][s] : NEG_;
      mx = fmaxf(mx, sc[s]);
    }
    float sm = 0.f;
    for (int s = 0; s < MAXSPAN_; s++) { sc[s] = __expf(sc[s] - mx); sm += sc[s]; }
    for (int s = 0; s < MAXSPAN_; s++) sw[s] = sc[s] / sm;
  }
  __syncthreads();
#pragma unroll
  for (int k2 = 0; k2 < 3; k2++) {
    int d = tid + k2 * 256;
    float pooled = 0.f;
#pragma unroll
    for (int s = 0; s < MAXSPAN_; s++) pooled += sw[s] * ch[s][k2];
    float outv = (len > 1) ? pooled : ch[0][k2];
    if (!valid) outv = wv[k2];
    if (l >= frst) outv = emb[(size_t)brkw[b] * D_ + d];
    size_t o = ((size_t)b * T_ + l) * D_ + d;
    fuse[o] = outv;
    fuse_h[o] = (_Float16)outv;
  }
}

__global__ void k_gather_h(const int* __restrict__ wi, const float* __restrict__ emb,
                           _Float16* __restrict__ dst) {
  int i = blockIdx.x * blockDim.x + threadIdx.x;
  if (i >= B_ * L_ * D_) return;
  int d = i % D_, bl = i / D_;
  dst[i] = (_Float16)emb[(size_t)wi[bl * 3] * D_ + d];
}

// ---------------------------------------------------------------------------
// NT GEMM, software-pipelined (ping-pong, unroll 2) K loop.
// mode 0: f32 store + bias; mode 1: tanh(f32)+bias; mode 2: f16 store + bias
// block = 128 thr = 4 waves; wave tile 32x64 (2 A-frags x 4 B-frags);
// block tile 32x256.
// ---------------------------------------------------------------------------
__global__ void __launch_bounds__(128) k_gemm_nt(const _Float16* __restrict__ A,
                                                 const _Float16* __restrict__ Bm,
                                                 const float* __restrict__ bias,
                                                 float* __restrict__ C,
                                                 _Float16* __restrict__ Ch, int M,
                                                 int N, int Kd, int mode) {
  int wave = threadIdx.x >> 5;
  int m0 = blockIdx.x * 32;
  int n0 = blockIdx.y * 256 + wave * 64;
  if (m0 >= M || n0 >= N) return;
  f8 acc[2][4];
#pragma unroll
  for (int rt = 0; rt < 2; rt++)
#pragma unroll
    for (int j = 0; j < 4; j++)
#pragma unroll
      for (int e = 0; e < 8; e++) acc[rt][j][e] = 0.f;

  const _Float16* Ab0 = A + (size_t)m0 * Kd;
  const _Float16* Ab1 = Ab0 + (size_t)16 * Kd;
  const _Float16* Bb[4];
#pragma unroll
  for (int j = 0; j < 4; j++) Bb[j] = Bm + (size_t)(n0 + j * 16) * Kd;

  // prologue fragments (k=0)
  h16 a0c = load_frag_nt(Ab0, Kd, 0);
  h16 a1c = load_frag_nt(Ab1, Kd, 0);
  h16 bfc[4];
#pragma unroll
  for (int j = 0; j < 4; j++) bfc[j] = load_frag_nt(Bb[j], Kd, 0);

#pragma unroll 2
  for (int k0 = 32; k0 < Kd; k0 += 32) {
    h16 a0n = load_frag_nt(Ab0, Kd, k0);    // prefetch next-k fragments
    h16 a1n = load_frag_nt(Ab1, Kd, k0);
    h16 bfn[4];
#pragma unroll
    for (int j = 0; j < 4; j++) bfn[j] = load_frag_nt(Bb[j], Kd, k0);
#pragma unroll
    for (int j = 0; j < 4; j++) {
      acc[0][j] = wmma_f16(a0c, bfc[j], acc[0][j]);
      acc[1][j] = wmma_f16(a1c, bfc[j], acc[1][j]);
    }
    a0c = a0n;
    a1c = a1n;
#pragma unroll
    for (int j = 0; j < 4; j++) bfc[j] = bfn[j];
  }
#pragma unroll
  for (int j = 0; j < 4; j++) {
    acc[0][j] = wmma_f16(a0c, bfc[j], acc[0][j]);
    acc[1][j] = wmma_f16(a1c, bfc[j], acc[1][j]);
  }

  int lane = threadIdx.x & 31;
  int col = lane & 15;
  int rofs = (lane >> 4) * 8;
#pragma unroll
  for (int rt = 0; rt < 2; rt++) {
#pragma unroll
    for (int j = 0; j < 4; j++) {
      int n = n0 + j * 16 + col;
      float bv = bias ? bias[n] : 0.f;
#pragma unroll
      for (int r = 0; r < 8; r++) {
        float v = acc[rt][j][r] + bv;
        size_t o = (size_t)(m0 + rt * 16 + r + rofs) * N + n;
        if (mode == 2) Ch[o] = (_Float16)v;
        else C[o] = (mode == 1) ? tanhf(v) : v;
      }
    }
  }
}

// ---------------------------------------------------------------------------
// dim attention: scores, softmax over words, attention + mask mix
// ---------------------------------------------------------------------------
__global__ void k_scores(const float* __restrict__ U, const float* __restrict__ emb,
                         const int* __restrict__ di, float* __restrict__ sc) {
  int gw = blockIdx.x * (blockDim.x >> 5) + (threadIdx.x >> 5);
  if (gw >= B_ * NDIM_ * L_) return;
  int lane = threadIdx.x & 31;
  int l = gw % L_, n = (gw / L_) % NDIM_, b = gw / (L_ * NDIM_);
  const float* u = U + ((size_t)b * L_ + l) * D_;
  const float* de = emb + (size_t)di[b * NDIM_ + n] * D_;
  float s = 0.f;
  for (int d = lane; d < D_; d += 32) s += u[d] * de[d];
#pragma unroll
  for (int o = 16; o; o >>= 1) s += __shfl_down(s, o, 32);
  if (!lane) sc[((size_t)b * NDIM_ + n) * L_ + l] = s;
}

__global__ void k_softmaxL(float* __restrict__ sc) {
  int gw = blockIdx.x * (blockDim.x >> 5) + (threadIdx.x >> 5);
  if (gw >= B_ * NDIM_) return;
  int lane = threadIdx.x & 31;
  float* row = sc + (size_t)gw * L_;
  float v[10], mx = -3.4e38f;
#pragma unroll
  for (int i = 0; i < 10; i++) {
    int l = lane + i * 32;
    v[i] = (l < L_) ? row[l] : -3.4e38f;
    mx = fmaxf(mx, v[i]);
  }
#pragma unroll
  for (int o = 16; o; o >>= 1) mx = fmaxf(mx, __shfl_xor(mx, o, 32));
  float sm = 0.f;
#pragma unroll
  for (int i = 0; i < 10; i++) {
    int l = lane + i * 32;
    if (l < L_) { v[i] = __expf(v[i] - mx); sm += v[i]; }
  }
#pragma unroll
  for (int o = 16; o; o >>= 1) sm += __shfl_xor(sm, o, 32);
#pragma unroll
  for (int i = 0; i < 10; i++) {
    int l = lane + i * 32;
    if (l < L_) row[l] = v[i] / sm;
  }
}

__global__ void k_attn(const float* __restrict__ alpha, const int* __restrict__ wi,
                       const float* __restrict__ emb, const int* __restrict__ mask,
                       float* __restrict__ fuse, _Float16* __restrict__ fuse_h) {
  int i = blockIdx.x * blockDim.x + threadIdx.x;
  if (i >= B_ * NDIM_ * D_) return;
  int d = i % D_, n = (i / D_) % NDIM_, b = i / (D_ * NDIM_);
  const float* ar = alpha + ((size_t)b * NDIM_ + n) * L_;
  const int* w = wi + (size_t)b * L_ * 3;
  float s = 0.f;
  for (int l = 0; l < L_; l++) s += ar[l] * emb[(size_t)w[l * 3] * D_ + d];
  float m = (float)mask[b * NDIM_ + n];
  float v = s * (1.f - m) + emb[d] * m;
  size_t o = ((size_t)b * T_ + L_ + n) * D_ + d;
  fuse[o] = v;
  fuse_h[o] = (_Float16)v;
}

// ---------------------------------------------------------------------------
// persistent BiLSTM recurrence; grid=2 (dir), 768 thr = 24 waves.
// Wave w owns h-channels [16w,16w+16) for all 4 gates x both 16-row tiles.
// h in LDS f16, c in VGPRs. Precomputed f16 gate tiles (32 x 1536 per step)
// are staged into LDS by the Tensor Data Mover one step ahead, overlapped
// with the WMMA recurrence. Only 2 barriers per step: the s_wait_tensorcnt
// stalls wave 0 only and is folded before the first barrier.
// Dynamic LDS: hbuf 24KB + gate tile 96KB = 120KB.
// ---------------------------------------------------------------------------
__global__ void __launch_bounds__(768, 1) k_lstm(const _Float16* __restrict__ gx_all,
                                                 const _Float16* __restrict__ Whh_all,
                                                 const float* __restrict__ h0,
                                                 const float* __restrict__ c0,
                                                 float* __restrict__ out) {
  extern __shared__ char smem[];
  _Float16* hbuf = (_Float16*)smem;                       // 32*384 f16
  _Float16* gxb  = (_Float16*)(smem + B_ * H_ * 2);       // 32*1536 f16

  int dir = blockIdx.x;
  const _Float16* gx  = gx_all + (size_t)dir * B_ * T_ * G4_;
  const _Float16* Whh = Whh_all + (size_t)dir * G4_ * H_;
  int tid = threadIdx.x;
  for (int i = tid; i < B_ * H_; i += 768)
    hbuf[i] = (_Float16)h0[(size_t)dir * B_ * H_ + i];

  int lane = tid & 31;
  int ht = tid >> 5;  // 0..23
  int col = lane & 15;
  int rofs = (lane >> 4) * 8;

  unsigned gx_lds = (unsigned)(size_t)(void*)gxb;  // LDS byte offset (low 32 bits)

  const _Float16* wb[4];
#pragma unroll
  for (int g = 0; g < 4; g++) wb[g] = Whh + (size_t)(g * H_ + ht * 16) * H_;

  f8 acc[2][4], creg[2];
#pragma unroll
  for (int rt = 0; rt < 2; rt++)
#pragma unroll
    for (int r = 0; r < 8; r++) {
      int b = rt * 16 + rofs + r;
      creg[rt][r] = c0[(size_t)dir * B_ * H_ + (size_t)b * H_ + ht * 16 + col];
    }

  // TDM: stage gate tile for step 0
  if (tid < 32) {
    int t0 = dir ? (T_ - 1) : 0;
    tdm_load_2d_f16(gx_lds, gx + (size_t)t0 * G4_, G4_, B_, G4_, B_,
                    (unsigned long long)T_ * G4_);
  }

  // B fragments for k=0 (Whh is time-invariant; pipeline wraps to k=0 so these
  // stay primed across timesteps)
  h16 bfc[4];
#pragma unroll
  for (int g = 0; g < 4; g++) bfc[g] = load_frag_nt(wb[g], H_, 0);

  __syncthreads();

  for (int s = 0; s < T_; s++) {
    int t = dir ? (T_ - 1 - s) : s;
#pragma unroll
    for (int rt = 0; rt < 2; rt++)
#pragma unroll
      for (int g = 0; g < 4; g++)
#pragma unroll
        for (int e = 0; e < 8; e++) acc[rt][g][e] = 0.f;

#pragma unroll 2
    for (int k0 = 0; k0 < H_; k0 += 32) {
      h16 af0 = load_frag_nt(hbuf, H_, k0);
      h16 af1 = load_frag_nt(hbuf + 16 * H_, H_, k0);
      int kn = (k0 + 32 < H_) ? (k0 + 32) : 0;  // wrap: prefetch next step's k=0
      h16 bfn[4];
#pragma unroll
      for (int g = 0; g < 4; g++) bfn[g] = load_frag_nt(wb[g], H_, kn);
#pragma unroll
      for (int g = 0; g < 4; g++) {
        acc[0][g] = wmma_f16(af0, bfc[g], acc[0][g]);
        acc[1][g] = wmma_f16(af1, bfc[g], acc[1][g]);
      }
#pragma unroll
      for (int g = 0; g < 4; g++) bfc[g] = bfn[g];
    }
    if (tid < 32) wait_tensor0();  // gate tile for step s resident (wave0 only)
    __syncthreads();               // h(t) reads done + gate tile visible to all

#pragma unroll
    for (int rt = 0; rt < 2; rt++) {
#pragma unroll
      for (int r = 0; r < 8; r++) {
        int b = rt * 16 + rofs + r;
        int hh = ht * 16 + col;
        const _Float16* grow = gxb + b * G4_;
        float gi = acc[rt][0][r] + (float)grow[0 * H_ + hh];
        float gf = acc[rt][1][r] + (float)grow[1 * H_ + hh];
        float gg = acc[rt][2][r] + (float)grow[2 * H_ + hh];
        float go = acc[rt][3][r] + (float)grow[3 * H_ + hh];
        float c = sigmoidf_(gf) * creg[rt][r] + sigmoidf_(gi) * tanhf(gg);
        float h = sigmoidf_(go) * tanhf(c);
        creg[rt][r] = c;
        hbuf[b * H_ + hh] = (_Float16)h;
        out[((size_t)b * T_ + t) * (2 * H_) + dir * H_ + hh] = h;
      }
    }
    __syncthreads();          // gxb fully consumed; h(t+1) published
    if (s + 1 < T_ && tid < 32) {
      int tn = dir ? (T_ - 2 - s) : (s + 1);
      tdm_load_2d_f16(gx_lds, gx + (size_t)tn * G4_, G4_, B_, G4_, B_,
                      (unsigned long long)T_ * G4_);   // overlaps next GEMM
    }
  }
}

// ---------------------------------------------------------------------------
// top-384-of-768 mean per (b,t): bitonic sort (pad to 1024) in LDS
// ---------------------------------------------------------------------------
__global__ void k_topk(const float* __restrict__ lo, float* __restrict__ pooled) {
  __shared__ float arr[1024];
  __shared__ float red[256];
  int row = blockIdx.x, tid = threadIdx.x;
  const float* src = lo + (size_t)row * (2 * H_);
  for (int i = tid; i < 1024; i += 256) arr[i] = (i < 2 * H_) ? src[i] : -3.4e38f;
  __syncthreads();
  for (int k = 2; k <= 1024; k <<= 1) {
    for (int j = k >> 1; j > 0; j >>= 1) {
      for (int i = tid; i < 1024; i += 256) {
        int ixj = i ^ j;
        if (ixj > i) {
          float a = arr[i], b = arr[ixj];
          bool up = ((i & k) == 0);
          if ((a > b) == up) { arr[i] = b; arr[ixj] = a; }
        }
      }
      __syncthreads();
    }
  }
  float s = 0.f;
  for (int i = 1024 - KTOP_ + tid; i < 1024; i += 256) s += arr[i];
  red[tid] = s;
  __syncthreads();
  for (int st = 128; st; st >>= 1) {
    if (tid < st) red[tid] += red[tid + st];
    __syncthreads();
  }
  if (!tid) pooled[row] = red[0] / (float)KTOP_;
}

__global__ void k_head(const float* __restrict__ pooled, const float* __restrict__ hW,
                       const float* __restrict__ hb, float* __restrict__ out) {
  __shared__ float pl[T_];
  __shared__ float lg[NHEAD_ * NCLS_];
  int b = blockIdx.x, tid = threadIdx.x;
  for (int i = tid; i < T_; i += blockDim.x) pl[i] = pooled[(size_t)b * T_ + i];
  __syncthreads();
  if (tid < NHEAD_ * NCLS_) {
    const float* w = hW + (size_t)tid * T_;
    float s = hb[tid];
    for (int n = 0; n < T_; n++) s += pl[n] * w[n];
    lg[tid] = s;
  }
  __syncthreads();
  if (tid < NHEAD_) {
    float mx = -3.4e38f;
    for (int o = 0; o < NCLS_; o++) mx = fmaxf(mx, lg[tid * NCLS_ + o]);
    float e[NCLS_], sm = 0.f;
    for (int o = 0; o < NCLS_; o++) { e[o] = __expf(lg[tid * NCLS_ + o] - mx); sm += e[o]; }
    for (int o = 0; o < NCLS_; o++)
      out[(size_t)b * NHEAD_ * NCLS_ + tid * NCLS_ + o] = e[o] / sm;
  }
}

// ---------------------------------------------------------------------------
extern "C" void kernel_launch(void* const* d_in, const int* in_sizes, int n_in,
                              void* d_out, int out_size, void* d_ws, size_t ws_size,
                              hipStream_t stream) {
  const float* ernie  = (const float*)d_in[0];
  const int* word_idx = (const int*)d_in[1];
  const int* dim_idx  = (const int*)d_in[2];
  const int* dim_mask = (const int*)d_in[3];
  const float* emb    = (const float*)d_in[4];
  const float* attn_W = (const float*)d_in[5];
  const float* attn_b = (const float*)d_in[6];
  const float* Wih    = (const float*)d_in[7];
  const float* Whh    = (const float*)d_in[8];
  const float* bih    = (const float*)d_in[9];
  const float* bhh    = (const float*)d_in[10];
  const float* head_W = (const float*)d_in[11];
  const float* head_b = (const float*)d_in[12];
  const float* h0     = (const float*)d_in[13];
  const float* c0     = (const float*)d_in[14];

  char* base = (char*)d_ws;
  size_t off = 0;
  auto carve = [&](size_t bytes) -> void* {
    void* p = base + off;
    off += (bytes + 255) & ~(size_t)255;
    return p;
  };
  float*    fuse      = (float*)carve((size_t)B_ * T_ * D_ * 4);
  _Float16* fuse_h    = (_Float16*)carve((size_t)B_ * T_ * D_ * 2);
  _Float16* wordemb_h = (_Float16*)carve((size_t)B_ * L_ * D_ * 2);
  float*    U         = (float*)carve((size_t)B_ * L_ * D_ * 4);
  float*    sc        = (float*)carve((size_t)B_ * NDIM_ * L_ * 4);
  _Float16* gx_h      = (_Float16*)carve((size_t)2 * B_ * T_ * G4_ * 2);
  float*    lstm_out  = (float*)carve((size_t)B_ * T_ * 2 * H_ * 4);
  float*    pooled    = (float*)carve((size_t)B_ * T_ * 4);
  _Float16* attnW_h   = (_Float16*)carve((size_t)D_ * D_ * 2);
  _Float16* Wih_h     = (_Float16*)carve((size_t)2 * G4_ * D_ * 2);
  _Float16* Whh_h     = (_Float16*)carve((size_t)2 * G4_ * H_ * 2);
  float*    biassum   = (float*)carve((size_t)2 * G4_ * 4);
  int*      firstb    = (int*)carve((size_t)B_ * 4);
  int*      brkw      = (int*)carve((size_t)B_ * 4);

  const int LSTM_LDS = (B_ * H_ + B_ * G4_) * 2;  // 24KB + 96KB
  hipFuncSetAttribute((void*)k_lstm, hipFuncAttributeMaxDynamicSharedMemorySize,
                      LSTM_LDS);

  // weight conversions + fused bias
  k_cvt_f16<<<(D_ * D_ + 255) / 256, 256, 0, stream>>>(attn_W, attnW_h, D_ * D_);
  k_cvt_f16<<<(2 * G4_ * D_ + 255) / 256, 256, 0, stream>>>(Wih, Wih_h, 2 * G4_ * D_);
  k_cvt_f16<<<(2 * G4_ * H_ + 255) / 256, 256, 0, stream>>>(Whh, Whh_h, 2 * G4_ * H_);
  k_add2<<<(2 * G4_ + 255) / 256, 256, 0, stream>>>(bih, bhh, biassum, 2 * G4_);

  // weighted embedding -> fuse[:, 0:300]
  k_first<<<1, 32, 0, stream>>>(word_idx, firstb, brkw);
  k_wemb<<<B_ * L_, 256, 0, stream>>>(ernie, word_idx, emb, firstb, brkw, fuse, fuse_h);

  // dim attention -> fuse[:, 300:305]
  k_gather_h<<<(B_ * L_ * D_) / 256, 256, 0, stream>>>(word_idx, emb, wordemb_h);
  k_gemm_nt<<<dim3((B_ * L_) / 32, D_ / 256), 128, 0, stream>>>(
      wordemb_h, attnW_h, attn_b, U, (_Float16*)nullptr, B_ * L_, D_, D_, 1);
  k_scores<<<(B_ * NDIM_ * L_) / 8, 256, 0, stream>>>(U, emb, dim_idx, sc);
  k_softmaxL<<<(B_ * NDIM_ + 7) / 8, 256, 0, stream>>>(sc);
  k_attn<<<(B_ * NDIM_ * D_) / 256, 256, 0, stream>>>(sc, word_idx, emb, dim_mask, fuse,
                                                      fuse_h);

  // time-parallel input projection per direction (bias folded), f16 output
  k_gemm_nt<<<dim3((B_ * T_) / 32, G4_ / 256), 128, 0, stream>>>(
      fuse_h, Wih_h, biassum, (float*)nullptr, gx_h, B_ * T_, G4_, D_, 2);
  k_gemm_nt<<<dim3((B_ * T_) / 32, G4_ / 256), 128, 0, stream>>>(
      fuse_h, Wih_h + (size_t)G4_ * D_, biassum + G4_,
      (float*)nullptr, gx_h + (size_t)B_ * T_ * G4_, B_ * T_, G4_, D_, 2);

  // sequential recurrence, one persistent workgroup per direction, TDM-staged
  k_lstm<<<2, 768, LSTM_LDS, stream>>>(gx_h, Whh_h, h0, c0, lstm_out);

  // top-k mean + head
  k_topk<<<B_ * T_, 256, 0, stream>>>(lstm_out, pooled);
  k_head<<<B_, 128, 0, stream>>>(pooled, head_W, head_b, (float*)d_out);
}
